// MultiHeadSelfAttention_82686710383177
// MI455X (gfx1250) — matrix operand order repaired and checked by reading, not compiled, not run on hardware
//
#include <hip/hip_runtime.h>
#include <stdint.h>

// ---------------------------------------------------------------------------
// Types for CDNA5 WMMA (wave32): v_wmma_f32_16x16x32_bf16
// ---------------------------------------------------------------------------
typedef __attribute__((ext_vector_type(16))) __bf16 bf16x16;
typedef __attribute__((ext_vector_type(8)))  __bf16 bf16x8;
typedef __attribute__((ext_vector_type(8)))  float  f32x8;

#define WMMA_BF16(a, b, c) \
    __builtin_amdgcn_wmma_f32_16x16x32_bf16(false, (a), false, (b), (short)0, (c), false, false)

// ---------------------------------------------------------------------------
// CDNA5 async global->LDS copies (ASYNCcnt-tracked, no VGPR round trip).
// Generic pointers to LDS carry the LDS byte offset in their low 32 bits.
// ---------------------------------------------------------------------------
__device__ __forceinline__ void g2l_async_b128(void* lds, const void* g) {
    uint32_t l = (uint32_t)(uintptr_t)lds;
    asm volatile("global_load_async_to_lds_b128 %0, %1, off"
                 :: "v"(l), "v"(g) : "memory");
}
// copy 32 bytes: two b128 with shared base + offset immediate
__device__ __forceinline__ void g2l_async_b256(void* lds, const void* g) {
    uint32_t l = (uint32_t)(uintptr_t)lds;
    asm volatile("global_load_async_to_lds_b128 %0, %1, off\n\t"
                 "global_load_async_to_lds_b128 %0, %1, off offset:16"
                 :: "v"(l), "v"(g) : "memory");
}
__device__ __forceinline__ void wait_async0() {
    asm volatile("s_wait_asynccnt 0x0" ::: "memory");
}

// ---------------------------------------------------------------------------
// Fragment loaders (wave32; A/B use contiguous k order, C per ISA 7.12.2)
//  A 16x32 bf16: lane l -> row m = l&15; half h = l>>4;
//    vector elems 0..15 = K = h*16 + 0..15      (contiguous)
//  B 32x16 bf16: lane l -> col n = l&15; half h = l>>4; elem e -> K = h*16 + e
//  C 16x16 f32 : lane l -> col n = l&15; VGPR r -> row m = r + 8*(l>>4)
// ---------------------------------------------------------------------------
__device__ __forceinline__ bf16x16 frag_a_load(const __bf16* base, int ld) {
    const int lane = threadIdx.x & 31;
    const int m = lane & 15, hh = lane >> 4;
    const __bf16* p = base + m * ld + hh * 16;
    union { bf16x16 v; bf16x8 h[2]; } u;
    u.h[0] = *(const bf16x8*)(p);
    u.h[1] = *(const bf16x8*)(p + 8);
    return u.v;
}

// B stored transposed in LDS as [n][k] (n rows, k contiguous) -> 2x16B loads
__device__ __forceinline__ bf16x16 frag_bt_load(const __bf16* baseT, int ld) {
    const int lane = threadIdx.x & 31;
    const int n = lane & 15, hh = lane >> 4;
    const __bf16* p = baseT + n * ld + hh * 16;
    union { bf16x16 v; bf16x8 h[2]; } u;
    u.h[0] = *(const bf16x8*)(p);
    u.h[1] = *(const bf16x8*)(p + 8);
    return u.v;
}

// B stored natural [k][n] in LDS (used for V tile): strided gather
__device__ __forceinline__ bf16x16 frag_b_load(const __bf16* base, int ld) {
    const int lane = threadIdx.x & 31;
    const int n = lane & 15, hh = lane >> 4;
    union { bf16x16 v; __bf16 e[16]; } u;
#pragma unroll
    for (int e = 0; e < 16; ++e) u.e[e] = base[(hh * 16 + e) * ld + n];
    return u.v;
}

// ---------------------------------------------------------------------------
// fp32 -> bf16 conversion
// ---------------------------------------------------------------------------
__global__ void cvt_f32_bf16(const float* __restrict__ in, __bf16* __restrict__ out, int n) {
    int i = blockIdx.x * blockDim.x + threadIdx.x;
    if (i < n) out[i] = (__bf16)in[i];
}

// ---------------------------------------------------------------------------
// Y = X @ W^T  (X: RxK bf16 row-major, W: NxK bf16 row-major)
// Block tile 128x128, K-step 32, 8 waves (4 along M x 2 along N), wave 32x64.
// Double-buffered LDS staged with async global->LDS copies.
// Output: bf16 (Ybf) or f32 (Yf).
// ---------------------------------------------------------------------------
#define GLDA 40
#define GLDB 40

__global__ void __launch_bounds__(256)
gemm_xwT(const __bf16* __restrict__ X, const __bf16* __restrict__ W,
         __bf16* __restrict__ Ybf, float* __restrict__ Yf,
         int R, int N, int K) {
    __shared__ __align__(16) __bf16 As[2][128][GLDA];
    __shared__ __align__(16) __bf16 Bs[2][128][GLDB];

    const int tid = threadIdx.x;
    const int wid = tid >> 5;
    const int lane = tid & 31;
    const int wm = wid & 3;       // wave M tile: wm*32
    const int wn = wid >> 2;      // wave N tile: wn*64
    const int rowBase = blockIdx.x * 128;
    const int colBase = blockIdx.y * 128;

    f32x8 acc[2][4];
#pragma unroll
    for (int mt = 0; mt < 2; ++mt)
#pragma unroll
        for (int nt = 0; nt < 4; ++nt) {
            f32x8 z = {0.f, 0.f, 0.f, 0.f, 0.f, 0.f, 0.f, 0.f};
            acc[mt][nt] = z;
        }

    const int lr = tid >> 1;            // 0..127 (row within tile)
    const int lc = (tid & 1) * 16;      // 0 or 16 (col within K-step)

    const __bf16* gx0 = X + (size_t)(rowBase + lr) * K + lc;
    const __bf16* gw0 = W + (size_t)(colBase + lr) * K + lc;

    // stage first K-step into buffer 0
    g2l_async_b256(&As[0][lr][lc], gx0);
    g2l_async_b256(&Bs[0][lr][lc], gw0);

    int buf = 0;
    for (int k0 = 0; k0 < K; k0 += 32) {
        wait_async0();          // our share of buffer `buf` has landed in LDS
        __syncthreads();        // everyone's share has landed

        if (k0 + 32 < K) {      // stage next K-step while we compute this one
            g2l_async_b256(&As[buf ^ 1][lr][lc], gx0 + k0 + 32);
            g2l_async_b256(&Bs[buf ^ 1][lr][lc], gw0 + k0 + 32);
        }

        bf16x16 af[2], bfr[4];
#pragma unroll
        for (int mt = 0; mt < 2; ++mt)
            af[mt] = frag_a_load(&As[buf][wm * 32 + mt * 16][0], GLDA);
#pragma unroll
        for (int nt = 0; nt < 4; ++nt)
            bfr[nt] = frag_bt_load(&Bs[buf][wn * 64 + nt * 16][0], GLDB);
#pragma unroll
        for (int mt = 0; mt < 2; ++mt)
#pragma unroll
            for (int nt = 0; nt < 4; ++nt)
                acc[mt][nt] = WMMA_BF16(af[mt], bfr[nt], acc[mt][nt]);

        __syncthreads();        // done reading `buf` before it is restaged
        buf ^= 1;
    }

    const int nl = lane & 15, hh = lane >> 4;
#pragma unroll
    for (int mt = 0; mt < 2; ++mt)
#pragma unroll
        for (int nt = 0; nt < 4; ++nt) {
            const float* cv = (const float*)&acc[mt][nt];
#pragma unroll
            for (int r = 0; r < 8; ++r) {
                int row = rowBase + wm * 32 + mt * 16 + r + 8 * hh;
                int col = colBase + wn * 64 + nt * 16 + nl;
                if (Ybf) Ybf[(size_t)row * N + col] = (__bf16)cv[r];
                else     Yf [(size_t)row * N + col] = cv[r];
            }
        }
}

// ---------------------------------------------------------------------------
// RoPE applied in place to Q and K (bf16, layout [B*S][H*dk])
// ---------------------------------------------------------------------------
__global__ void rope_kernel(__bf16* __restrict__ Q, __bf16* __restrict__ Kb,
                            int BS, int S, int D, int dk) {
    int i = blockIdx.x * blockDim.x + threadIdx.x;
    int half = D >> 1;
    if (i >= BS * half) return;
    int row = i / half, p = i % half;
    int head = p / (dk >> 1);
    int j = p % (dk >> 1);
    int pos = row % S;
    float inv = powf(10000.0f, -2.0f * (float)j / (float)dk);
    float ang = (float)pos * inv;
    float c = cosf(ang), s = sinf(ang);
    size_t base = (size_t)row * D + (size_t)head * dk + 2 * j;
    float q1 = (float)Q[base], q2 = (float)Q[base + 1];
    Q[base]     = (__bf16)(q1 * c - q2 * s);
    Q[base + 1] = (__bf16)(q1 * s + q2 * c);
    float k1 = (float)Kb[base], k2 = (float)Kb[base + 1];
    Kb[base]     = (__bf16)(k1 * c - k2 * s);
    Kb[base + 1] = (__bf16)(k1 * s + k2 * c);
}

// ---------------------------------------------------------------------------
// Flash attention: block = 64 query rows of one (b,h); 4 waves x 16 rows.
// QK^T and P.V on WMMA; online softmax in f32 with lane-butterfly reductions.
// K/V tiles double-buffered and staged with async global->LDS copies so the
// next tile streams in while the current tile's WMMAs + softmax execute.
// ---------------------------------------------------------------------------
#define ALD 72

__global__ void __launch_bounds__(128)
attn_kernel(const __bf16* __restrict__ Q, const __bf16* __restrict__ K,
            const __bf16* __restrict__ V, __bf16* __restrict__ O,
            int B, int S, int H, int dk) {
    const int qb = blockIdx.x;                 // query tile (64 rows)
    const int bh = blockIdx.y;
    const int b = bh / H, hd = bh % H;
    const int D = H * dk;

    const int tid = threadIdx.x;
    const int w = tid >> 5;
    const int lane = tid & 31;
    const int nl = lane & 15, hh = lane >> 4;

    __shared__ __align__(16) __bf16 Qs[64][ALD];
    __shared__ __align__(16) __bf16 Ks[2][64][ALD];
    __shared__ __align__(16) __bf16 Vs[2][64][ALD];
    __shared__ __align__(16) __bf16 Ps[4][16][ALD];

    const size_t bS = (size_t)b * S;
    const size_t hoff = (size_t)hd * dk;

    // stage Q tile (async; waited together with first K/V stage)
    const size_t qrow0 = bS + (size_t)qb * 64;
    for (int i = tid; i < 64 * 8; i += 128) {
        int r = i >> 3, c = (i & 7) * 8;
        g2l_async_b128(&Qs[r][c], Q + (qrow0 + r) * D + hoff + c);
    }
    // stage first K/V tile into buffer 0
    for (int i = tid; i < 64 * 8; i += 128) {
        int r = i >> 3, c = (i & 7) * 8;
        const size_t goff = (bS + r) * D + hoff + c;   // kb = 0
        g2l_async_b128(&Ks[0][r][c], K + goff);
        g2l_async_b128(&Vs[0][r][c], V + goff);
    }

    f32x8 o[4];
    float m_run[8], l_run[8];
#pragma unroll
    for (int d = 0; d < 4; ++d) {
        f32x8 z = {0.f, 0.f, 0.f, 0.f, 0.f, 0.f, 0.f, 0.f};
        o[d] = z;
    }
#pragma unroll
    for (int r = 0; r < 8; ++r) { m_run[r] = -1e30f; l_run[r] = 0.f; }

    const float scale = 0.125f;  // 1/sqrt(64)

    int buf = 0;
    for (int kb = 0; kb <= qb; ++kb) {
        wait_async0();                         // buffer `buf` (and Q) landed
        __syncthreads();

        if (kb + 1 <= qb) {                    // prefetch next K/V tile
            const size_t krow1 = bS + (size_t)(kb + 1) * 64;
            for (int i = tid; i < 64 * 8; i += 128) {
                int r = i >> 3, c = (i & 7) * 8;
                const size_t goff = (krow1 + r) * D + hoff + c;
                g2l_async_b128(&Ks[buf ^ 1][r][c], K + goff);
                g2l_async_b128(&Vs[buf ^ 1][r][c], V + goff);
            }
        }

        // ---- S = Q_strip(16x64) @ K^T(64x64) -> 4 C frags ----
        f32x8 sc[4];
#pragma unroll
        for (int nt = 0; nt < 4; ++nt) {
            f32x8 z = {0.f, 0.f, 0.f, 0.f, 0.f, 0.f, 0.f, 0.f};
            sc[nt] = z;
        }
#pragma unroll
        for (int k0 = 0; k0 < 64; k0 += 32) {
            bf16x16 a = frag_a_load(&Qs[w * 16][k0], ALD);
#pragma unroll
            for (int nt = 0; nt < 4; ++nt) {
                bf16x16 bb = frag_bt_load(&Ks[buf][nt * 16][k0], ALD);
                sc[nt] = WMMA_BF16(a, bb, sc[nt]);
            }
        }

        // ---- scale + causal mask ----
        const int qloc = w * 16;  // this wave's first query row within tile
#pragma unroll
        for (int nt = 0; nt < 4; ++nt) {
            float* s = (float*)&sc[nt];
#pragma unroll
            for (int r = 0; r < 8; ++r) {
                s[r] *= scale;
                if (kb == qb && (nt * 16 + nl) > (qloc + r + 8 * hh)) s[r] = -1e30f;
            }
        }

        // ---- online softmax (row stats over 64 columns) ----
        float alpha[8];
#pragma unroll
        for (int r = 0; r < 8; ++r) {
            float mx = ((float*)&sc[0])[r];
#pragma unroll
            for (int nt = 1; nt < 4; ++nt) mx = fmaxf(mx, ((float*)&sc[nt])[r]);
            mx = fmaxf(mx, __shfl_xor(mx, 1, 32));
            mx = fmaxf(mx, __shfl_xor(mx, 2, 32));
            mx = fmaxf(mx, __shfl_xor(mx, 4, 32));
            mx = fmaxf(mx, __shfl_xor(mx, 8, 32));
            float mnew = fmaxf(m_run[r], mx);
            alpha[r] = __expf(m_run[r] - mnew);
            m_run[r] = mnew;
            float rs = 0.f;
#pragma unroll
            for (int nt = 0; nt < 4; ++nt) {
                float* s = (float*)&sc[nt];
                float pv = __expf(s[r] - mnew);
                s[r] = pv;
                rs += pv;
            }
            rs += __shfl_xor(rs, 1, 32);
            rs += __shfl_xor(rs, 2, 32);
            rs += __shfl_xor(rs, 4, 32);
            rs += __shfl_xor(rs, 8, 32);
            l_run[r] = l_run[r] * alpha[r] + rs;
        }

        // ---- stash P (C layout -> row-major bf16 in per-wave LDS) ----
#pragma unroll
        for (int nt = 0; nt < 4; ++nt) {
            const float* s = (const float*)&sc[nt];
#pragma unroll
            for (int r = 0; r < 8; ++r)
                Ps[w][r + 8 * hh][nt * 16 + nl] = (__bf16)s[r];
        }

        // ---- rescale O, then O += P(16x64) @ V(64x64) ----
#pragma unroll
        for (int d = 0; d < 4; ++d) {
            float* ov = (float*)&o[d];
#pragma unroll
            for (int r = 0; r < 8; ++r) ov[r] *= alpha[r];
        }
#pragma unroll
        for (int k0 = 0; k0 < 64; k0 += 32) {
            bf16x16 a = frag_a_load(&Ps[w][0][k0], ALD);
#pragma unroll
            for (int d = 0; d < 4; ++d) {
                bf16x16 bb = frag_b_load(&Vs[buf][k0][d * 16], ALD);
                o[d] = WMMA_BF16(a, bb, o[d]);
            }
        }

        __syncthreads();       // all waves done reading `buf` before restage
        buf ^= 1;
    }

    // ---- epilogue: normalize and store context (bf16) ----
#pragma unroll
    for (int d = 0; d < 4; ++d) {
        const float* ov = (const float*)&o[d];
#pragma unroll
        for (int r = 0; r < 8; ++r) {
            int row = qb * 64 + w * 16 + r + 8 * hh;
            float val = ov[r] / l_run[r];
            O[(bS + row) * D + hoff + d * 16 + nl] = (__bf16)val;
        }
    }
}

// ---------------------------------------------------------------------------
// Host-side orchestration
// ---------------------------------------------------------------------------
extern "C" void kernel_launch(void* const* d_in, const int* in_sizes, int n_in,
                              void* d_out, int out_size, void* d_ws, size_t ws_size,
                              hipStream_t stream) {
    const float* x  = (const float*)d_in[0];
    const float* Wq = (const float*)d_in[1];
    const float* Wk = (const float*)d_in[2];
    const float* Wv = (const float*)d_in[3];
    const float* Wo = (const float*)d_in[4];

    const int S = 2048, H = 16, dk = 64;
    const int D = 1024;
    const int R = in_sizes[0] / D;            // B*S = 4096
    const int B = R / S;                      // 2

    char* ws = (char*)d_ws;
    __bf16* xb   = (__bf16*)ws; ws += (size_t)R * D * sizeof(__bf16);
    __bf16* wqb  = (__bf16*)ws; ws += (size_t)D * D * sizeof(__bf16);
    __bf16* wkb  = (__bf16*)ws; ws += (size_t)D * D * sizeof(__bf16);
    __bf16* wvb  = (__bf16*)ws; ws += (size_t)D * D * sizeof(__bf16);
    __bf16* wob  = (__bf16*)ws; ws += (size_t)D * D * sizeof(__bf16);
    __bf16* qbuf = (__bf16*)ws; ws += (size_t)R * D * sizeof(__bf16);
    __bf16* kbuf = (__bf16*)ws; ws += (size_t)R * D * sizeof(__bf16);
    __bf16* vbuf = (__bf16*)ws; ws += (size_t)R * D * sizeof(__bf16);
    __bf16* ctx  = (__bf16*)ws; ws += (size_t)R * D * sizeof(__bf16);

    const int nXD = R * D, nWW = D * D;
    cvt_f32_bf16<<<(nXD + 255) / 256, 256, 0, stream>>>(x,  xb,  nXD);
    cvt_f32_bf16<<<(nWW + 255) / 256, 256, 0, stream>>>(Wq, wqb, nWW);
    cvt_f32_bf16<<<(nWW + 255) / 256, 256, 0, stream>>>(Wk, wkb, nWW);
    cvt_f32_bf16<<<(nWW + 255) / 256, 256, 0, stream>>>(Wv, wvb, nWW);
    cvt_f32_bf16<<<(nWW + 255) / 256, 256, 0, stream>>>(Wo, wob, nWW);

    dim3 gg(R / 128, D / 128);
    gemm_xwT<<<gg, 256, 0, stream>>>(xb, wqb, qbuf, (float*)nullptr, R, D, D);
    gemm_xwT<<<gg, 256, 0, stream>>>(xb, wkb, kbuf, (float*)nullptr, R, D, D);
    gemm_xwT<<<gg, 256, 0, stream>>>(xb, wvb, vbuf, (float*)nullptr, R, D, D);

    int nPairs = R * (D / 2);
    rope_kernel<<<(nPairs + 255) / 256, 256, 0, stream>>>(qbuf, kbuf, R, S, D, dk);

    dim3 ga(S / 64, B * H);
    attn_kernel<<<ga, 128, 0, stream>>>(qbuf, kbuf, vbuf, ctx, B, S, H, dk);

    gemm_xwT<<<gg, 256, 0, stream>>>(ctx, wob, (__bf16*)nullptr, (float*)d_out, R, D, D);
}
